// QuantLinearInt8_51814485459239
// MI455X (gfx1250) — 8-bit matrix operand order repaired and checked by reading, compiled, not run on hardware
//
#include <hip/hip_runtime.h>
#include <stdint.h>

typedef __attribute__((ext_vector_type(8))) int   v8i;
typedef __attribute__((ext_vector_type(4))) float v4f;

__global__ __launch_bounds__(256) void quant_rows_kernel(
    const float* __restrict__ x, int8_t* __restrict__ xq,
    float* __restrict__ xscale, int IN)
{
    const int t = blockIdx.x;
    const v4f* __restrict__ xr = (const v4f*)(x + (size_t)t * IN);
    const int n4 = IN >> 2;
    const int chunks = n4 / blockDim.x;

    v4f cache[8];
    float amax = 0.0f;
#pragma unroll
    for (int c = 0; c < 8; ++c) {
        if (c < chunks) {
            v4f v = __builtin_nontemporal_load(&xr[threadIdx.x + c * blockDim.x]);
            cache[c] = v;
            amax = fmaxf(amax, fmaxf(fmaxf(fabsf(v[0]), fabsf(v[1])),
                                     fmaxf(fabsf(v[2]), fabsf(v[3]))));
        }
    }
    for (int i = threadIdx.x + 8 * blockDim.x; i < n4; i += blockDim.x) {
        v4f v = xr[i];
        amax = fmaxf(amax, fmaxf(fmaxf(fabsf(v[0]), fabsf(v[1])),
                                 fmaxf(fabsf(v[2]), fabsf(v[3]))));
    }

    for (int off = 16; off > 0; off >>= 1)
        amax = fmaxf(amax, __shfl_xor(amax, off, 32));

    __shared__ float sred[8];
    const int lane = threadIdx.x & 31;
    const int wv   = threadIdx.x >> 5;
    if (lane == 0) sred[wv] = amax;
    __syncthreads();
    if (wv == 0) {
        float m = (lane < 8) ? sred[lane] : 0.0f;
        for (int off = 4; off > 0; off >>= 1)
            m = fmaxf(m, __shfl_xor(m, off, 32));
        if (lane == 0) sred[0] = fmaxf(m, 1e-8f) / 127.0f;
    }
    __syncthreads();
    const float scale = sred[0];
    if (threadIdx.x == 0) xscale[t] = scale;

    const float inv = 1.0f / scale;
    uint32_t* __restrict__ qr = (uint32_t*)(xq + (size_t)t * IN);
#pragma unroll
    for (int c = 0; c < 8; ++c) {
        if (c < chunks) {
            v4f v = cache[c];
            int a0 = (int)fminf(fmaxf(rintf(v[0] * inv), -127.0f), 127.0f);
            int a1 = (int)fminf(fmaxf(rintf(v[1] * inv), -127.0f), 127.0f);
            int a2 = (int)fminf(fmaxf(rintf(v[2] * inv), -127.0f), 127.0f);
            int a3 = (int)fminf(fmaxf(rintf(v[3] * inv), -127.0f), 127.0f);
            uint32_t packed = (uint32_t)(a0 & 0xFF) | ((uint32_t)(a1 & 0xFF) << 8) |
                              ((uint32_t)(a2 & 0xFF) << 16) | ((uint32_t)(a3 & 0xFF) << 24);
            qr[threadIdx.x + c * blockDim.x] = packed;
        }
    }
    for (int i = threadIdx.x + 8 * blockDim.x; i < n4; i += blockDim.x) {
        v4f v = xr[i];
        int a0 = (int)fminf(fmaxf(rintf(v[0] * inv), -127.0f), 127.0f);
        int a1 = (int)fminf(fmaxf(rintf(v[1] * inv), -127.0f), 127.0f);
        int a2 = (int)fminf(fmaxf(rintf(v[2] * inv), -127.0f), 127.0f);
        int a3 = (int)fminf(fmaxf(rintf(v[3] * inv), -127.0f), 127.0f);
        qr[i] = (uint32_t)(a0 & 0xFF) | ((uint32_t)(a1 & 0xFF) << 8) |
                ((uint32_t)(a2 & 0xFF) << 16) | ((uint32_t)(a3 & 0xFF) << 24);
    }
}

__global__ __launch_bounds__(256) void gemm_int8_wmma_kernel(
    const int8_t* __restrict__ xq, const int8_t* __restrict__ wgt,
    const float* __restrict__ xscale, const float* __restrict__ wscale,
    const float* __restrict__ bias, float* __restrict__ out,
    int IN, int OUT)
{
    const int lane = threadIdx.x & 31;
    const int wave = threadIdx.x >> 5;
    const int wm = wave & 1;
    const int wn = wave >> 1;
    const int l15 = lane & 15;
    const int lh  = lane >> 4;

    const int rowBlock = blockIdx.x * 128;
    const int colBlock = blockIdx.y * 128;

    const int8_t* aBase = xq  + (size_t)(rowBlock + wm * 64 + l15) * IN + lh * 16;
    const int8_t* bBase = wgt + (size_t)(colBlock + wn * 32 + l15) * IN + lh * 16;

    v8i acc[4][2] = {};

    for (int k = 0; k < IN; k += 64) {
        v8i a[4];
#pragma unroll
        for (int mt = 0; mt < 4; ++mt) {
            const int8_t* p = aBase + (size_t)mt * 16 * IN + k;
            int2 q0 = *(const int2*)(p);
            int2 q1 = *(const int2*)(p + 8);
            int2 q2 = *(const int2*)(p + 32);
            int2 q3 = *(const int2*)(p + 40);
            a[mt][0] = q0.x; a[mt][1] = q0.y;
            a[mt][2] = q1.x; a[mt][3] = q1.y;
            a[mt][4] = q2.x; a[mt][5] = q2.y;
            a[mt][6] = q3.x; a[mt][7] = q3.y;
        }
        v8i b[2];
#pragma unroll
        for (int nt = 0; nt < 2; ++nt) {
            const int8_t* p = bBase + (size_t)nt * 16 * IN + k;
            int4 r0 = *(const int4*)(p);
            int4 r1 = *(const int4*)(p + 32);
            b[nt][0] = r0.x; b[nt][1] = r0.y; b[nt][2] = r0.z; b[nt][3] = r0.w;
            b[nt][4] = r1.x; b[nt][5] = r1.y; b[nt][6] = r1.z; b[nt][7] = r1.w;
        }
#pragma unroll
        for (int mt = 0; mt < 4; ++mt)
#pragma unroll
            for (int nt = 0; nt < 2; ++nt)
                acc[mt][nt] = __builtin_amdgcn_wmma_i32_16x16x64_iu8(
                     true, a[mt],  true, b[nt],
                    acc[mt][nt],  false,  false);
    }

#pragma unroll
    for (int nt = 0; nt < 2; ++nt) {
        const int col = colBlock + wn * 32 + nt * 16 + l15;
        const float wsc = wscale[col];
        const float bs  = bias[col];
#pragma unroll
        for (int mt = 0; mt < 4; ++mt) {
            const int rbase = rowBlock + wm * 64 + mt * 16 + lh * 8;
            v8i a = acc[mt][nt];
#pragma unroll
            for (int r = 0; r < 8; ++r) {
                const int row = rbase + r;
                float val = (float)a[r] * xscale[row] * wsc + bs;
                __builtin_nontemporal_store(val, &out[(size_t)row * OUT + col]);
            }
        }
    }
}

extern "C" void kernel_launch(void* const* d_in, const int* in_sizes, int n_in,
                              void* d_out, int out_size, void* d_ws, size_t ws_size,
                              hipStream_t stream)
{
    const float*  x      = (const float*)d_in[0];
    const int8_t* wgt    = (const int8_t*)d_in[1];
    const float*  wscale = (const float*)d_in[2];
    const float*  bias   = (const float*)d_in[3];
    float*        out    = (float*)d_out;

    const int OUT = in_sizes[3];
    const int IN  = in_sizes[1] / OUT;
    const int T   = in_sizes[0] / IN;

    int8_t* xq     = (int8_t*)d_ws;
    float*  xscale = (float*)((char*)d_ws + (size_t)T * IN);

    quant_rows_kernel<<<T, 256, 0, stream>>>(x, xq, xscale, IN);

    dim3 grid(T / 128, OUT / 128);
    gemm_int8_wmma_kernel<<<grid, dim3(256), 0, stream>>>(
        xq, wgt, xscale, wscale, bias, out, IN, OUT);
}
